// PropNet_34308198760590
// MI455X (gfx1250) — compile-verified, matrix operand order checked by reading, not run on hardware
//
#include <hip/hip_runtime.h>
#include <hip/hip_bf16.h>

// PropNet (interaction network) for MI455X / gfx1250.
// B=4, N=1024, E=8192, F_in=32, H=128, out=3, pstep=2.
// Strategy: recover gather/scatter indices from the dense one-hot Rr/Rs
// (one coalesced pass), then run all MLP layers as f16 WMMA GEMMs with fp32
// accumulation (v_wmma_f32_16x16x32_f16), fused bias+residual+ReLU epilogue.

#define B_   4
#define N_   1024
#define E_   8192
#define H_   128
#define FIN_ 32
#define PSTEP_ 2   // reference passes pstep=2 (python scalar)

typedef _Float16 f16;
typedef __attribute__((ext_vector_type(16))) _Float16 v16h;
typedef __attribute__((ext_vector_type(8)))  _Float16 v8h;
typedef __attribute__((ext_vector_type(8)))  float    v8f;

// ---------------------------------------------------------------------------
// WMMA fragment load, 16-bit A/B (B loaded column-major from W^T).
// ISA layout (16x32 f16 A): lanes 0-15 = rows 0-15 holding K {0..7,16..23},
// lanes 16-31 = rows 0-15 holding K {8..15,24..31}. Two 16B loads per frag.
// ---------------------------------------------------------------------------
__device__ __forceinline__ v16h load_frag(const f16* __restrict__ base,
                                          int stride, int lane) {
  const int hi16 = lane >> 4;          // 0 or 1
  const int r    = lane & 15;          // row (A) / col (B^T row)
  const f16* p = base + r * stride + (hi16 << 3);
  union { v16h v; v8h h[2]; } u;
  u.h[0] = *(const v8h*)(p);           // K = kbase .. kbase+7
  u.h[1] = *(const v8h*)(p + 16);      // K = kbase+16 .. kbase+23
  return u.v;
}

// ---------------------------------------------------------------------------
// Generic GEMM: out = act(A[M,K] @ W[K,128] + bias (+resid)), W given as
// Wt[128,K] = W^T in f16. One wave per 16x16 C tile; 4 waves / block.
// grid = (M/64, 8), block = 128.
// ---------------------------------------------------------------------------
__global__ void k_gemm_wmma(const f16*  __restrict__ A,
                            const f16*  __restrict__ Wt,
                            const float* __restrict__ bias,
                            const float* __restrict__ resid,   // may be null
                            f16*   __restrict__ out16,         // may be null
                            float* __restrict__ out32,         // may be null
                            int M, int K, int do_relu) {
  const int lane  = threadIdx.x & 31;
  const int wave  = threadIdx.x >> 5;
  const int mtile = blockIdx.x * 4 + wave;
  const int ntile = blockIdx.y;                 // 0..7 (Nout = 128)
  if (mtile * 16 >= M) return;                  // wave-uniform; EXEC stays full

  const int col = ntile * 16 + (lane & 15);
  const float bv = bias[col];
  v8f c;
#pragma unroll
  for (int v = 0; v < 8; ++v) c[v] = bv;

  const f16* Ab = A  + (long)mtile * 16 * K;
  const f16* Bb = Wt + (long)ntile * 16 * K;
  for (int kk = 0; kk < K; kk += 32) {
    v16h a = load_frag(Ab + kk, K, lane);
    v16h b = load_frag(Bb + kk, K, lane);
    c = __builtin_amdgcn_wmma_f32_16x16x32_f16(false, a, false, b,
                                               (short)0, c, false, false);
  }

  // C layout: lanes 0-15 -> rows v, lanes 16-31 -> rows v+8; col = lane&15.
  const int rbase = mtile * 16 + ((lane >> 4) << 3);
#pragma unroll
  for (int v = 0; v < 8; ++v) {
    const int idx = (rbase + v) * H_ + col;
    float val = c[v];
    if (resid)   val += resid[idx];
    if (do_relu) val  = fmaxf(val, 0.0f);
    if (out32)   out32[idx] = val;
    if (out16)   out16[idx] = (f16)val;
  }
}

// --------------------------- elementwise helpers ---------------------------
__global__ void k_f32_to_f16(const float* __restrict__ in, f16* __restrict__ out, int n) {
  for (int i = blockIdx.x * blockDim.x + threadIdx.x; i < n; i += gridDim.x * blockDim.x)
    out[i] = (f16)in[i];
}

__global__ void k_copy_f16(const f16* __restrict__ in, f16* __restrict__ out, int n) {
  for (int i = blockIdx.x * blockDim.x + threadIdx.x; i < n; i += gridDim.x * blockDim.x)
    out[i] = in[i];
}

__global__ void k_zero_f32(float* __restrict__ p, int n) {
  for (int i = blockIdx.x * blockDim.x + threadIdx.x; i < n; i += gridDim.x * blockDim.x)
    p[i] = 0.0f;
}

// W[K,128] (f32) -> Wt[128,K] (f16)
__global__ void k_wt(const float* __restrict__ W, f16* __restrict__ Wt, int K) {
  const int total = K * H_;
  for (int i = blockIdx.x * blockDim.x + threadIdx.x; i < total; i += gridDim.x * blockDim.x) {
    const int k = i / H_, n = i % H_;
    Wt[n * K + k] = (f16)W[i];
  }
}

// Recover indices from dense one-hot rows: exactly one element > 0.5 per row.
__global__ void k_onehot_idx(const float* __restrict__ R, int* __restrict__ idx, long total) {
  for (long i = (long)blockIdx.x * blockDim.x + threadIdx.x; i < total;
       i += (long)gridDim.x * blockDim.x)
    if (R[i] > 0.5f) idx[i / N_] = (int)(i % N_);
}

// Edge encoder input: [x_r | x_s], 2*FIN_ per edge.
__global__ void k_build_edge_x(const f16* __restrict__ x16,
                               const int* __restrict__ recv, const int* __restrict__ send,
                               f16* __restrict__ out) {
  const int total = B_ * E_ * (2 * FIN_);
  for (int i = blockIdx.x * blockDim.x + threadIdx.x; i < total; i += gridDim.x * blockDim.x) {
    const int be = i >> 6, j = i & 63;
    const int b  = be / E_;
    const int node = (j < FIN_) ? recv[be] : send[be];
    out[i] = x16[(b * N_ + node) * FIN_ + (j & (FIN_ - 1))];
  }
}

// Edge step input: [edge_encode | ne_r | ne_s], 3*H_ per edge.
__global__ void k_build_edge_in(const f16* __restrict__ eenc, const f16* __restrict__ neff,
                                const int* __restrict__ recv, const int* __restrict__ send,
                                f16* __restrict__ out) {
  const int total = B_ * E_ * (3 * H_);
  for (int i = blockIdx.x * blockDim.x + threadIdx.x; i < total; i += gridDim.x * blockDim.x) {
    const int be = i / (3 * H_), j = i - be * (3 * H_);
    const int b  = be / E_;
    f16 v;
    if (j < H_)            v = eenc[be * H_ + j];
    else if (j < 2 * H_)   v = neff[(b * N_ + recv[be]) * H_ + (j - H_)];
    else                   v = neff[(b * N_ + send[be]) * H_ + (j - 2 * H_)];
    out[i] = v;
  }
}

// agg[b, recv[e], h] += edge_effect[b, e, h]
__global__ void k_scatter_add(const float* __restrict__ eff,
                              const int* __restrict__ recv,
                              float* __restrict__ agg) {
  const int total = B_ * E_ * H_;
  for (int i = blockIdx.x * blockDim.x + threadIdx.x; i < total; i += gridDim.x * blockDim.x) {
    const int be = i >> 7, h = i & (H_ - 1);
    const int b  = be / E_;
    atomicAdd(&agg[(b * N_ + recv[be]) * H_ + h], eff[i]);
  }
}

// [a(f16,128) | b(f32,128)] -> out f16[rows,256]
__global__ void k_concat_f16_f32(const f16* __restrict__ a, const float* __restrict__ b,
                                 f16* __restrict__ out, int rows) {
  const int total = rows * 2 * H_;
  for (int i = blockIdx.x * blockDim.x + threadIdx.x; i < total; i += gridDim.x * blockDim.x) {
    const int r = i >> 8, j = i & 255;
    out[i] = (j < H_) ? a[r * H_ + j] : (f16)b[r * H_ + (j - H_)];
  }
}

// [a(f16,128) | b(f16,128)] -> out f16[rows,256]
__global__ void k_concat_f16_f16(const f16* __restrict__ a, const f16* __restrict__ b,
                                 f16* __restrict__ out, int rows) {
  const int total = rows * 2 * H_;
  for (int i = blockIdx.x * blockDim.x + threadIdx.x; i < total; i += gridDim.x * blockDim.x) {
    const int r = i >> 8, j = i & 255;
    out[i] = (j < H_) ? a[r * H_ + j] : b[r * H_ + (j - H_)];
  }
}

// Final 128 -> 3 layer, fp32 weights, no ReLU.
__global__ void k_final3(const f16* __restrict__ h, const float* __restrict__ W,
                         const float* __restrict__ b, float* __restrict__ out, int M) {
  const int i = blockIdx.x * blockDim.x + threadIdx.x;
  if (i >= M * 3) return;
  const int m = i / 3, j = i - m * 3;
  float acc = b[j];
  const f16* hp = h + m * H_;
#pragma unroll 8
  for (int k = 0; k < H_; ++k) acc += (float)hp[k] * W[k * 3 + j];
  out[i] = acc;
}

// ---------------------------------------------------------------------------
extern "C" void kernel_launch(void* const* d_in, const int* in_sizes, int n_in,
                              void* d_out, int out_size, void* d_ws, size_t ws_size,
                              hipStream_t stream) {
  const float* x    = (const float*)d_in[0];
  const float* Rr   = (const float*)d_in[1];
  const float* Rs   = (const float*)d_in[2];
  // d_in[3] = pstep (== 2 in reference; loop count must be host-static for capture)
  const float *neW0 = (const float*)d_in[4],  *neb0 = (const float*)d_in[5];
  const float *neW1 = (const float*)d_in[6],  *neb1 = (const float*)d_in[7];
  const float *neW2 = (const float*)d_in[8],  *neb2 = (const float*)d_in[9];
  const float *eeW0 = (const float*)d_in[10], *eeb0 = (const float*)d_in[11];
  const float *eeW1 = (const float*)d_in[12], *eeb1 = (const float*)d_in[13];
  const float *eeW2 = (const float*)d_in[14], *eeb2 = (const float*)d_in[15];
  const float *npW  = (const float*)d_in[16], *npb  = (const float*)d_in[17];
  const float *epW  = (const float*)d_in[18], *epb  = (const float*)d_in[19];
  const float *ppW0 = (const float*)d_in[20], *ppb0 = (const float*)d_in[21];
  const float *ppW1 = (const float*)d_in[22], *ppb1 = (const float*)d_in[23];
  const float *ppW2 = (const float*)d_in[24], *ppb2 = (const float*)d_in[25];
  float* out = (float*)d_out;

  // ---- workspace bump allocator (256B aligned) ----
  char* ws = (char*)d_ws;
  size_t off = 0;
  auto alloc = [&](size_t bytes) -> void* {
    void* p = ws + off;
    off = (off + bytes + 255) & ~(size_t)255;
    return p;
  };

  f16* x16    = (f16*)alloc((size_t)B_ * N_ * FIN_ * 2);
  int* recv   = (int*)alloc((size_t)B_ * E_ * 4);
  int* send   = (int*)alloc((size_t)B_ * E_ * 4);
  f16* neW0t  = (f16*)alloc((size_t)FIN_   * H_ * 2);
  f16* neW1t  = (f16*)alloc((size_t)H_     * H_ * 2);
  f16* neW2t  = (f16*)alloc((size_t)H_     * H_ * 2);
  f16* eeW0t  = (f16*)alloc((size_t)2*FIN_ * H_ * 2);
  f16* eeW1t  = (f16*)alloc((size_t)H_     * H_ * 2);
  f16* eeW2t  = (f16*)alloc((size_t)H_     * H_ * 2);
  f16* npWt   = (f16*)alloc((size_t)2*H_   * H_ * 2);
  f16* epWt   = (f16*)alloc((size_t)3*H_   * H_ * 2);
  f16* ppW0t  = (f16*)alloc((size_t)2*H_   * H_ * 2);
  f16* ppW1t  = (f16*)alloc((size_t)H_     * H_ * 2);
  f16* xe16   = (f16*)alloc((size_t)B_ * E_ * 2*FIN_ * 2);
  f16* eh16a  = (f16*)alloc((size_t)B_ * E_ * H_ * 2);
  f16* eh16b  = (f16*)alloc((size_t)B_ * E_ * H_ * 2);
  f16* eenc16 = (f16*)alloc((size_t)B_ * E_ * H_ * 2);
  f16* ein16  = (f16*)alloc((size_t)B_ * E_ * 3*H_ * 2);
  float* eeff32 = (float*)alloc((size_t)B_ * E_ * H_ * 4);
  float* agg32  = (float*)alloc((size_t)B_ * N_ * H_ * 4);
  f16* nh16a  = (f16*)alloc((size_t)B_ * N_ * H_ * 2);
  f16* nh16b  = (f16*)alloc((size_t)B_ * N_ * H_ * 2);
  f16* nenc16 = (f16*)alloc((size_t)B_ * N_ * H_ * 2);
  f16* neff16 = (f16*)alloc((size_t)B_ * N_ * H_ * 2);
  float* neff32 = (float*)alloc((size_t)B_ * N_ * H_ * 4);
  f16* nin16  = (f16*)alloc((size_t)B_ * N_ * 2*H_ * 2);
  f16* pin16  = (f16*)alloc((size_t)B_ * N_ * 2*H_ * 2);
  f16* ph16a  = (f16*)alloc((size_t)B_ * N_ * H_ * 2);
  f16* ph16b  = (f16*)alloc((size_t)B_ * N_ * H_ * 2);
  (void)ws_size; (void)n_in; (void)in_sizes; (void)out_size;

  const int TB = 256;
  auto blk = [](long n, int t) { return (unsigned)((n + t - 1) / t); };
  const int MN = B_ * N_;   // 4096
  const int ME = B_ * E_;   // 32768
  const dim3 gN(MN / 64, 8), gE(ME / 64, 8), bG(128);

  // ---- index extraction + precision prep ----
  k_f32_to_f16<<<blk(MN * FIN_, TB), TB, 0, stream>>>(x, x16, MN * FIN_);
  k_onehot_idx<<<65535, TB, 0, stream>>>(Rr, recv, (long)B_ * E_ * N_);
  k_onehot_idx<<<65535, TB, 0, stream>>>(Rs, send, (long)B_ * E_ * N_);
  k_wt<<<blk(FIN_ * H_, TB), TB, 0, stream>>>(neW0, neW0t, FIN_);
  k_wt<<<blk(H_ * H_, TB), TB, 0, stream>>>(neW1, neW1t, H_);
  k_wt<<<blk(H_ * H_, TB), TB, 0, stream>>>(neW2, neW2t, H_);
  k_wt<<<blk(2*FIN_ * H_, TB), TB, 0, stream>>>(eeW0, eeW0t, 2*FIN_);
  k_wt<<<blk(H_ * H_, TB), TB, 0, stream>>>(eeW1, eeW1t, H_);
  k_wt<<<blk(H_ * H_, TB), TB, 0, stream>>>(eeW2, eeW2t, H_);
  k_wt<<<blk(2*H_ * H_, TB), TB, 0, stream>>>(npW, npWt, 2*H_);
  k_wt<<<blk(3*H_ * H_, TB), TB, 0, stream>>>(epW, epWt, 3*H_);
  k_wt<<<blk(2*H_ * H_, TB), TB, 0, stream>>>(ppW0, ppW0t, 2*H_);
  k_wt<<<blk(H_ * H_, TB), TB, 0, stream>>>(ppW1, ppW1t, H_);

  // ---- node encoder: 32 -> 128 -> 128 -> 128 (ReLU all) ----
  k_gemm_wmma<<<gN, bG, 0, stream>>>(x16,  neW0t, neb0, nullptr, nh16a, nullptr, MN, FIN_, 1);
  k_gemm_wmma<<<gN, bG, 0, stream>>>(nh16a, neW1t, neb1, nullptr, nh16b, nullptr, MN, H_, 1);
  k_gemm_wmma<<<gN, bG, 0, stream>>>(nh16b, neW2t, neb2, nullptr, nenc16, neff32, MN, H_, 1);
  k_copy_f16<<<blk(MN * H_, TB), TB, 0, stream>>>(nenc16, neff16, MN * H_);

  // ---- edge encoder: 64 -> 128 -> 128 -> 128 (ReLU all) ----
  k_build_edge_x<<<blk((long)ME * 2*FIN_, TB), TB, 0, stream>>>(x16, recv, send, xe16);
  k_gemm_wmma<<<gE, bG, 0, stream>>>(xe16,  eeW0t, eeb0, nullptr, eh16a, nullptr, ME, 2*FIN_, 1);
  k_gemm_wmma<<<gE, bG, 0, stream>>>(eh16a, eeW1t, eeb1, nullptr, eh16b, nullptr, ME, H_, 1);
  k_gemm_wmma<<<gE, bG, 0, stream>>>(eh16b, eeW2t, eeb2, nullptr, eenc16, nullptr, ME, H_, 1);

  // ---- propagation (pstep = 2) ----
  for (int step = 0; step < PSTEP_; ++step) {
    k_build_edge_in<<<blk((long)ME * 3*H_, TB), TB, 0, stream>>>(eenc16, neff16, recv, send, ein16);
    k_gemm_wmma<<<gE, bG, 0, stream>>>(ein16, epWt, epb, nullptr, nullptr, eeff32, ME, 3*H_, 1);
    k_zero_f32<<<blk(MN * H_, TB), TB, 0, stream>>>(agg32, MN * H_);
    k_scatter_add<<<blk((long)ME * H_, TB), TB, 0, stream>>>(eeff32, recv, agg32);
    k_concat_f16_f32<<<blk(MN * 2*H_, TB), TB, 0, stream>>>(nenc16, agg32, nin16, MN);
    k_gemm_wmma<<<gN, bG, 0, stream>>>(nin16, npWt, npb, neff32, neff16, neff32, MN, 2*H_, 1);
  }

  // ---- predictor: 256 -> 128 -> 128 -> 3 ----
  k_concat_f16_f16<<<blk(MN * 2*H_, TB), TB, 0, stream>>>(nenc16, neff16, pin16, MN);
  k_gemm_wmma<<<gN, bG, 0, stream>>>(pin16, ppW0t, ppb0, nullptr, ph16a, nullptr, MN, 2*H_, 1);
  k_gemm_wmma<<<gN, bG, 0, stream>>>(ph16a, ppW1t, ppb1, nullptr, ph16b, nullptr, MN, H_, 1);
  k_final3<<<blk(MN * 3, TB), TB, 0, stream>>>(ph16b, ppW2, ppb2, out, MN);
}
